// RGCN_SR_45243185496740
// MI455X (gfx1250) — matrix operand order repaired and checked by reading, not compiled, not run on hardware
//
#include <hip/hip_runtime.h>
#include <hip/hip_bf16.h>
#include <math.h>

// ---------------------------------------------------------------------------
// RGCN-LSTM: h_t recurrence with graph-conv A@q injected into the cell state.
// Dominant cost: A@q per step ([2048,2048]x[2048,64] = 537 MFLOP/step, 82% of
// FLOPs). A converted to bf16 once (8 MB, L2-resident on 192MB L2) and the
// GEMM runs on v_wmma_f32_16x16x32_bf16 with fp32 accumulation.
//
// Step2 mapping: each wave owns an exclusive K-slice of 256 and produces ALL
// 4 column tiles from a single A fragment -> A is loaded exactly once per
// block per step, 1 A-load : 4 WMMAs. Epilogue operands (act, c) are staged
// into LDS with global_load_async_to_lds_b128 issued at kernel entry,
// overlapped with the GEMM, drained with s_wait_asynccnt.
// ---------------------------------------------------------------------------

typedef __attribute__((ext_vector_type(16))) __bf16   v16bf;
typedef __attribute__((ext_vector_type(8)))  float    v8f;
typedef __attribute__((ext_vector_type(4)))  unsigned u32x4;
typedef int v4i __attribute__((vector_size(16)));   // matches builtin param

#define AS1 __attribute__((address_space(1)))
#define AS3 __attribute__((address_space(3)))

#define NN 2048
#define TT 365
#define DD 16
#define HH 64
#define G4 256   // 4*H

// ---------------------------------------------------------------------------
// Prologue: A(f32) -> A(bf16) in workspace, zero h and c. Runs every call so
// the launch is deterministic (harness does not re-poison between replays).
// ---------------------------------------------------------------------------
__global__ __launch_bounds__(256) void rgcn_prologue(
    const float* __restrict__ A, __bf16* __restrict__ Abf,
    float* __restrict__ h, float* __restrict__ c)
{
    size_t i = (size_t)blockIdx.x * 256 + threadIdx.x;
    if (i < (size_t)NN * NN) Abf[i] = (__bf16)A[i];
    if (i < (size_t)NN * HH) { h[i] = 0.0f; c[i] = 0.0f; }
}

// ---------------------------------------------------------------------------
// Step 1: gates = sigma/tanh(x_t@W_ih + h@W_hh + bias), q = tanh(h@W_q + b_q).
// Small K (80 / 64): fp32 VALU with weight-column reuse across 8 rows/block
// (weight L2 traffic /8). qT stored transposed in bf16 so step2's WMMA
// B-operand loads are contiguous per lane.
// ---------------------------------------------------------------------------
#define S1_ROWS 8
__global__ __launch_bounds__(256) void rgcn_step1(
    const float* __restrict__ x,      // [N,T,D]
    const float* __restrict__ W_ih,   // [D,4H]
    const float* __restrict__ W_hh,   // [H,4H]
    const float* __restrict__ bias,   // [4H]
    const float* __restrict__ W_q,    // [H,H]
    const float* __restrict__ b_q,    // [H]
    const float* __restrict__ h,      // [N,H]   (h_{t-1})
    float*       __restrict__ act,    // [N,4H]  activated i|f|g|o
    __bf16*      __restrict__ qT,     // [H,N]   transposed, bf16
    int t)
{
    __shared__ float xs[S1_ROWS][DD];
    __shared__ float hs[S1_ROWS][HH];

    const int tid = threadIdx.x;
    const int n0  = blockIdx.x * S1_ROWS;

    if (tid < S1_ROWS * DD)
        xs[tid / DD][tid % DD] =
            x[((size_t)(n0 + tid / DD) * TT + t) * DD + (tid % DD)];
    for (int e = tid; e < S1_ROWS * HH; e += 256)
        hs[e / HH][e % HH] = h[(size_t)(n0 + e / HH) * HH + (e % HH)];
    __syncthreads();

    const int col = tid;                       // gate column 0..255
    float acc[S1_ROWS];
    #pragma unroll
    for (int r = 0; r < S1_ROWS; ++r) acc[r] = bias[col];

    for (int k = 0; k < DD; ++k) {             // x_t @ W_ih
        float wv = W_ih[k * G4 + col];
        #pragma unroll
        for (int r = 0; r < S1_ROWS; ++r) acc[r] += xs[r][k] * wv;
    }
    for (int k = 0; k < HH; ++k) {             // h @ W_hh
        float wv = W_hh[k * G4 + col];
        #pragma unroll
        for (int r = 0; r < S1_ROWS; ++r) acc[r] += hs[r][k] * wv;
    }
    const int sel = col >> 6;                  // 0:i 1:f 2:g 3:o
    #pragma unroll
    for (int r = 0; r < S1_ROWS; ++r) {
        float a = (sel == 2) ? tanhf(acc[r])
                             : 1.0f / (1.0f + __expf(-acc[r]));
        act[(size_t)(n0 + r) * G4 + col] = a;
    }

    if (col < HH) {                            // q = tanh(h @ W_q + b_q)
        float q[S1_ROWS];
        #pragma unroll
        for (int r = 0; r < S1_ROWS; ++r) q[r] = b_q[col];
        for (int k = 0; k < HH; ++k) {
            float wv = W_q[k * HH + col];
            #pragma unroll
            for (int r = 0; r < S1_ROWS; ++r) q[r] += hs[r][k] * wv;
        }
        #pragma unroll
        for (int r = 0; r < S1_ROWS; ++r)
            qT[(size_t)col * NN + (n0 + r)] = (__bf16)tanhf(q[r]);
    }
}

// ---------------------------------------------------------------------------
// Step 2: Aq = A_bf16 @ q  (WMMA bf16, fp32 accum), then fused cell update
//   c' = f*(c + Aq) + i*g ; h' = o*tanh(c') ; out[n,t] = h'.W_dense + b.
// 128 blocks x 8 waves. Block = 16-row strip. Wave w owns K-slice
// [w*256,(w+1)*256): 8 iterations, each = 1 A fragment -> 4 WMMAs (one per
// 16-col tile). 8-way K reduction via LDS. act/c staged via async-to-LDS.
// ---------------------------------------------------------------------------
__global__ __launch_bounds__(256) void rgcn_step2(
    const __bf16* __restrict__ Abf,    // [N,N] bf16
    const __bf16* __restrict__ qT,     // [H,N] bf16 (q transposed)
    const float*  __restrict__ act,    // [N,4H] i|f|g|o
    float*        __restrict__ h,      // [N,H] in: h_{t-1} rows overwritten
    float*        __restrict__ c,      // [N,H]
    const float*  __restrict__ W_dense,// [H]
    const float*  __restrict__ b_dense,// [1]
    float*        __restrict__ out,    // [N,T]
    int t)
{
    __shared__ float lds_part[8][4][16][16]; // per-wave K-slice partials, 32KB
    __shared__ float lds_act[16][G4];        // async-staged activations, 16KB
    __shared__ float lds_c[16][HH];          // async-staged cell state,   4KB
    __shared__ float lds_h[16][HH];          // new h strip for output dot

    const int tid  = threadIdx.x;
    const int wave = tid >> 5;
    const int lane = tid & 31;
    const int r0   = blockIdx.x * 16;

    const int m    = lane & 15;    // row (A) / col (B) within tile
    const int half = lane >> 4;    // lane-half selects K sub-range

    // ---- Kick off async staging of epilogue operands (overlaps the GEMM).
    // act strip: 16 rows x 256 f32 = 16KB contiguous; c strip: 4KB contiguous.
    {
        AS1 char* gact = (AS1 char*)(act + (size_t)r0 * G4);
        AS1 char* gc   = (AS1 char*)(c   + (size_t)r0 * HH);
        AS3 char* lact = (AS3 char*)&lds_act[0][0];
        AS3 char* lc   = (AS3 char*)&lds_c[0][0];
        #pragma unroll
        for (int e = 0; e < 4; ++e) {
            int ci = (tid + e * 256) * 16;   // byte chunk
            __builtin_amdgcn_global_load_async_to_lds_b128(
                (AS1 v4i*)(gact + ci), (AS3 v4i*)(lact + ci), 0, 0);
        }
        __builtin_amdgcn_global_load_async_to_lds_b128(
            (AS1 v4i*)(gc + tid * 16), (AS3 v4i*)(lc + tid * 16), 0, 0);
    }

    // A-operand row (row-major, stride N): contiguous K per lane half —
    // lane half 0: v0..7 = K[k0,k0+16)
    // lane half 1: v0..7 = K[k0+16,k0+32)
    const __bf16* arow = Abf + (size_t)(r0 + m) * NN;
    // B-operand: column (cj*16+m) of q == row of qT (contiguous K);
    // lane half selects K[k0,+16) vs K[k0+16,+32) (dense-B analog of the
    // ISA sparse-B layout: lanes 0-15 low K half, lanes 16-31 high K half).
    const __bf16* b0 = qT + (size_t)( 0 + m) * NN;
    const __bf16* b1 = qT + (size_t)(16 + m) * NN;
    const __bf16* b2 = qT + (size_t)(32 + m) * NN;
    const __bf16* b3 = qT + (size_t)(48 + m) * NN;

    v8f acc0 = {}, acc1 = {}, acc2 = {}, acc3 = {};
    const int kbase = wave * (NN / 8);       // exclusive K-slice per wave
    for (int k0 = kbase; k0 < kbase + NN / 8; k0 += 32) {
        union { u32x4 u[2]; v16bf v; } a, f0, f1, f2, f3;
        a.u[0]  = *(const u32x4*)(arow + k0 + half * 16);
        a.u[1]  = *(const u32x4*)(arow + k0 + half * 16 + 8);
        f0.u[0] = *(const u32x4*)(b0 + k0 + half * 16);
        f0.u[1] = *(const u32x4*)(b0 + k0 + half * 16 + 8);
        f1.u[0] = *(const u32x4*)(b1 + k0 + half * 16);
        f1.u[1] = *(const u32x4*)(b1 + k0 + half * 16 + 8);
        f2.u[0] = *(const u32x4*)(b2 + k0 + half * 16);
        f2.u[1] = *(const u32x4*)(b2 + k0 + half * 16 + 8);
        f3.u[0] = *(const u32x4*)(b3 + k0 + half * 16);
        f3.u[1] = *(const u32x4*)(b3 + k0 + half * 16 + 8);
        __builtin_prefetch(arow + k0 + 64, 0, 3);   // near-cache A prefetch
        acc0 = __builtin_amdgcn_wmma_f32_16x16x32_bf16(
                   false, a.v, false, f0.v, (short)0, acc0, false, false);
        acc1 = __builtin_amdgcn_wmma_f32_16x16x32_bf16(
                   false, a.v, false, f1.v, (short)0, acc1, false, false);
        acc2 = __builtin_amdgcn_wmma_f32_16x16x32_bf16(
                   false, a.v, false, f2.v, (short)0, acc2, false, false);
        acc3 = __builtin_amdgcn_wmma_f32_16x16x32_bf16(
                   false, a.v, false, f3.v, (short)0, acc3, false, false);
    }

    // C/D layout: VGPR v holds M = v + 8*half, lane&15 = N column.
    #pragma unroll
    for (int v = 0; v < 8; ++v) {
        int mrow = v + 8 * half;
        lds_part[wave][0][mrow][m] = acc0[v];
        lds_part[wave][1][mrow][m] = acc1[v];
        lds_part[wave][2][mrow][m] = acc2[v];
        lds_part[wave][3][mrow][m] = acc3[v];
    }

    // Drain async LDS copies issued by this wave, then rendezvous.
    asm volatile("s_wait_asynccnt 0x0" ::: "memory");
    __syncthreads();

    // Fused LSTM cell update: 16x64 elements, 4 per thread; 8-way K reduce.
    #pragma unroll
    for (int e = 0; e < 4; ++e) {
        int idx = tid + e * 256;      // 0..1023
        int mm  = idx >> 6;
        int j   = idx & 63;
        int n   = r0 + mm;
        float aq = 0.0f;
        #pragma unroll
        for (int w = 0; w < 8; ++w) aq += lds_part[w][j >> 4][mm][j & 15];
        float iv = lds_act[mm][j];
        float fv = lds_act[mm][64 + j];
        float gv = lds_act[mm][128 + j];
        float ov = lds_act[mm][192 + j];
        float cn = fv * (lds_c[mm][j] + aq) + iv * gv;
        float hn = ov * tanhf(cn);
        c[(size_t)n * HH + j] = cn;
        h[(size_t)n * HH + j] = hn;
        lds_h[mm][j] = hn;
    }
    __syncthreads();

    // out[n,t] = h_t . W_dense + b_dense  (never materialize [N,T,H])
    if (tid < 16) {
        int n = r0 + tid;
        float s = b_dense[0];
        #pragma unroll
        for (int j = 0; j < HH; ++j) s += lds_h[tid][j] * W_dense[j];
        out[(size_t)n * TT + t] = s;
    }
}

// ---------------------------------------------------------------------------
// Host launcher: prologue + 365 x (step1, step2), all on `stream`.
// ---------------------------------------------------------------------------
extern "C" void kernel_launch(void* const* d_in, const int* in_sizes, int n_in,
                              void* d_out, int out_size, void* d_ws, size_t ws_size,
                              hipStream_t stream)
{
    const float* x       = (const float*)d_in[0];
    const float* A       = (const float*)d_in[1];
    const float* W_ih    = (const float*)d_in[2];
    const float* W_hh    = (const float*)d_in[3];
    const float* bias    = (const float*)d_in[4];
    const float* W_q     = (const float*)d_in[5];
    const float* b_q     = (const float*)d_in[6];
    const float* W_dense = (const float*)d_in[7];
    const float* b_dense = (const float*)d_in[8];
    float* out = (float*)d_out;

    // Workspace layout (16B aligned slabs):
    //   Abf : N*N bf16 = 8 MiB | qT : H*N bf16 = 256 KiB
    //   act : N*4H f32 = 2 MiB | h,c : N*H f32 = 512 KiB each
    char* ws = (char*)d_ws;
    __bf16* Abf = (__bf16*)ws;
    __bf16* qT  = (__bf16*)(ws + (size_t)NN * NN * 2);
    float*  act = (float*)(ws + (size_t)NN * NN * 2 + (size_t)HH * NN * 2);
    float*  h   = act + (size_t)NN * G4;
    float*  c   = h   + (size_t)NN * HH;

    rgcn_prologue<<<(NN * NN + 255) / 256, 256, 0, stream>>>(A, Abf, h, c);

    for (int t = 0; t < TT; ++t) {
        rgcn_step1<<<NN / S1_ROWS, 256, 0, stream>>>(
            x, W_ih, W_hh, bias, W_q, b_q, h, act, qT, t);
        rgcn_step2<<<NN / 16, 256, 0, stream>>>(
            Abf, qT, act, h, c, W_dense, b_dense, out, t);
    }
}
